// SingleDecoderLayer_47974784697120
// MI455X (gfx1250) — matrix operand order repaired and checked by reading, not compiled, not run on hardware
//
#include <hip/hip_runtime.h>
#include <hip/hip_bf16.h>
#include <math.h>

// ---------------------------------------------------------------------------
// Problem constants (match reference)
// ---------------------------------------------------------------------------
#define BB   4
#define CIN  64
#define COUT 32
#define KS3  27          // 3*3*3 taps
#define MOD  128
#define BANK 4
#define DD   32
#define HH   32
#define WW   32
#define DHW  (DD*HH*WW)  // 32768
#define KTOT (CIN*KS3)   // 1728
#define NKS  (KTOT/32)   // 54 K-steps of 32
#define UP   64          // upsampled dim
#define UPVOX (UP*UP*UP) // 262144

typedef __attribute__((ext_vector_type(16))) _Float16 v16h;
typedef __attribute__((ext_vector_type(8)))  float    v8f;

// ---------------------------------------------------------------------------
// Kernel 1: style stats -> s[b][CIN], alpha[b][BANK] (softmax over 4)
// ---------------------------------------------------------------------------
__global__ void prep_stats_kernel(const float* __restrict__ style,
                                  const float* __restrict__ sel_w,
                                  const float* __restrict__ sel_b,
                                  const float* __restrict__ aff_w,
                                  const float* __restrict__ aff_b,
                                  float* __restrict__ s_out,
                                  float* __restrict__ alpha_out) {
  int b = blockIdx.x;
  int t = threadIdx.x;                    // 128 threads
  __shared__ float logits[BANK];
  if (t < CIN) {
    float acc = aff_b[t];
    for (int m = 0; m < MOD; ++m) acc += style[b*MOD + m] * aff_w[t*MOD + m];
    s_out[b*CIN + t] = acc;
  } else if (t < CIN + BANK) {
    int n = t - CIN;
    float acc = sel_b[n];
    for (int m = 0; m < MOD; ++m) acc += style[b*MOD + m] * sel_w[n*MOD + m];
    logits[n] = acc;
  }
  __syncthreads();
  if (t < BANK) {
    float mx = fmaxf(fmaxf(logits[0], logits[1]), fmaxf(logits[2], logits[3]));
    float e0 = expf(logits[0]-mx), e1 = expf(logits[1]-mx);
    float e2 = expf(logits[2]-mx), e3 = expf(logits[3]-mx);
    float inv = 1.0f / (e0+e1+e2+e3);
    alpha_out[b*BANK + t] = expf(logits[t]-mx) * inv;
  }
}

// ---------------------------------------------------------------------------
// Kernel 2: demodulation factor per (b, cout)
//   demod = rsqrt( sum_{cin,tap} (s[cin] * blend(cout,cin,tap))^2 + eps )
//   1728 = 64*27; iterate cin-major so tap = e & (31..) won't work (27 not
//   pow2) -> iterate (cin, tap) with explicit nests, threads stride over cin.
// ---------------------------------------------------------------------------
__global__ void prep_demod_kernel(const float* __restrict__ bank,
                                  const float* __restrict__ s_in,
                                  const float* __restrict__ alpha,
                                  float* __restrict__ demod_out) {
  int bc = blockIdx.x;                    // B*COUT blocks
  int b  = bc >> 5;
  int co = bc & 31;
  int t  = threadIdx.x;                   // 128 threads: t = cin*2 + half-tap
  float a0 = alpha[b*BANK+0], a1 = alpha[b*BANK+1];
  float a2 = alpha[b*BANK+2], a3 = alpha[b*BANK+3];
  const int NSTR = COUT*CIN*KS3;          // 55296 per bank entry
  int cin  = t >> 1;                      // 0..63
  int tap0 = (t & 1) ? 14 : 0;            // split 27 taps: 0..13 / 14..26
  int tapN = (t & 1) ? 27 : 14;
  float sc = s_in[b*CIN + cin];
  float acc = 0.0f;
  const float* bp = bank + ((size_t)co*CIN + cin)*KS3;
  for (int tap = tap0; tap < tapN; ++tap) {
    float wb = a0*bp[tap] + a1*bp[tap + NSTR]
             + a2*bp[tap + 2*NSTR] + a3*bp[tap + 3*NSTR];
    float w = wb * sc;
    acc += w * w;
  }
  __shared__ float red[128];
  red[t] = acc;
  __syncthreads();
  for (int off = 64; off > 0; off >>= 1) {
    if (t < off) red[t] += red[t + off];
    __syncthreads();
  }
  if (t == 0) demod_out[b*COUT + co] = rsqrtf(red[0] + 1e-8f);
}

// ---------------------------------------------------------------------------
// Kernel 3: write f16 weights pre-swizzled into WMMA A fragments (contiguous k).
//   16-bit A 16x32 (wave32):  lane 0-15  : M=lane,    K 0..15
//                             lane 16-31 : M=lane-16, K 16..31
//   VGPR v holds K pair: 2v,2v+1  (+16 hi lanes)
//   k ordering: k = tap*64 + cin   (CIN=64 -> tap = k>>6, cin = k&63)
//   afrag[b][mtile][ks][lane][16 halfs]  -> one aligned 32B vector per lane
// ---------------------------------------------------------------------------
__global__ void prep_afrag_kernel(const float* __restrict__ bank,
                                  const float* __restrict__ s_in,
                                  const float* __restrict__ alpha,
                                  const float* __restrict__ demod,
                                  _Float16* __restrict__ afrag) {
  int g    = blockIdx.x;                  // B*2*NKS blocks, 32 threads each
  int ks   = g % NKS;
  int mt   = (g / NKS) & 1;
  int b    = g / (2*NKS);
  int lane = threadIdx.x;
  int m    = lane & 15;
  int kb   = (lane < 16) ? 0 : 16;
  int co   = mt*16 + m;
  float a0 = alpha[b*BANK+0], a1 = alpha[b*BANK+1];
  float a2 = alpha[b*BANK+2], a3 = alpha[b*BANK+3];
  float dm = demod[b*COUT + co];
  const int NSTR = COUT*CIN*KS3;
  v16h vals;
  #pragma unroll
  for (int i = 0; i < 16; ++i) {
    int v = i >> 1, j = i & 1;
    int klocal = kb + i;
    int k   = ks*32 + klocal;
    int tap = k >> 6;
    int cin = k & 63;
    size_t idx = ((size_t)co*CIN + cin)*KS3 + tap;
    float wb = a0*bank[idx] + a1*bank[idx + NSTR]
             + a2*bank[idx + 2*NSTR] + a3*bank[idx + 3*NSTR];
    vals[i] = (_Float16)(wb * s_in[b*CIN + cin] * dm);
  }
  v16h* dst = (v16h*)afrag + ((size_t)(b*2 + mt)*NKS + ks)*32 + lane;
  *dst = vals;
}

// ---------------------------------------------------------------------------
// Kernel 4: implicit-GEMM conv via WMMA.
//   One block per (b, d, h) output row of 32 voxels, all 32 couts.
//   LDS slab: [dz][dy][w' 0..33][cin] f16 so each lane's B fragment
//   (16 consecutive cin) is one aligned 32B LDS vector.
//   w-halo columns (w'=0, w'=33) are ALWAYS out of range -> pure memset;
//   in-range plane region is 64x32 halfs -> all index math is shift/mask.
//   4 waves: (mtile, ntile) in {0,1}x{0,1}; K-loop = kd x kh x kw x cin-half
//   fully unrolled except kd -> 18 static v_wmma, no integer division.
// ---------------------------------------------------------------------------
#define SLAB (3*3*34*CIN)                 // 19584 halfs = 39168 B
__global__ __launch_bounds__(128) void conv_wmma_kernel(
    const float* __restrict__ x,
    const _Float16* __restrict__ afrag,
    float* __restrict__ y_conv) {
  __shared__ _Float16 lds[SLAB];
  int b = blockIdx.y;
  int d = blockIdx.x >> 5;
  int h = blockIdx.x & 31;
  int t = threadIdx.x;

  // ---- cooperative slab load (f32 global -> f16 LDS) ----
  #pragma unroll 1
  for (int dz = 0; dz < 3; ++dz) {
    #pragma unroll 1
    for (int dy = 0; dy < 3; ++dy) {
      int gd = d + dz - 1, gh = h + dy - 1;
      int lbase = (dz*3 + dy)*34*CIN;
      if (((unsigned)gd < 32u) && ((unsigned)gh < 32u)) {
        // w-halo columns: always zero (gw=-1 / gw=32)
        lds[lbase + ((t < 64) ? t : (33*CIN + (t - 64)))] = (_Float16)0.0f;
        const float* src = x + (((size_t)b*CIN)*DD + gd)*HH*WW + (size_t)gh*WW;
        #pragma unroll
        for (int j = 0; j < 16; ++j) {       // 2048 halfs = 16 * 128
          int f2  = j*128 + t;
          int cin = f2 >> 5;                 // 0..63
          int wl  = f2 & 31;                 // gw
          lds[lbase + (wl + 1)*CIN + cin] = (_Float16)src[(size_t)cin*DHW + wl];
        }
      } else {
        #pragma unroll
        for (int j = 0; j < 17; ++j) {       // 2176 halfs = 17 * 128
          int f2 = j*128 + t;
          if (f2 < 34*CIN) lds[lbase + f2] = (_Float16)0.0f;
        }
      }
    }
  }
  __syncthreads();

  // ---- WMMA main loop ----
  int wid  = t >> 5;
  int lane = t & 31;
  int mt   = wid & 1;
  int nt   = wid >> 1;
  int n    = lane & 15;
  int kb   = (lane < 16) ? 0 : 16;        // B layout: lo lanes K 0..15, hi 16..31

  v8f acc = {};
  const v16h* aptr = (const v16h*)afrag + (size_t)(b*2 + mt)*NKS*32 + lane;
  const _Float16* bcol = &lds[(nt*16 + n)*CIN + kb]; // w' = nt*16+n (+kw later)

  #pragma unroll 1
  for (int kd = 0; kd < 3; ++kd) {
    int ksbase = kd*18;                   // ks = tap*2 + ch, tap = (kd*3+kh)*3+kw
    if (kd < 2)                           // prefetch next kd plane of A frags
      __builtin_prefetch((const void*)(aptr + (size_t)(ksbase + 18)*32), 0, 1);
    #pragma unroll
    for (int kh = 0; kh < 3; ++kh) {
      const _Float16* bplane = bcol + (kd*3 + kh)*34*CIN;
      #pragma unroll
      for (int kw = 0; kw < 3; ++kw) {
        const _Float16* bline = bplane + kw*CIN;
        #pragma unroll
        for (int ch = 0; ch < 2; ++ch) {  // cin halves: cb = ch*32 + kb
          int idx = (kh*3 + kw)*2 + ch;
          v16h bm = *(const v16h*)(bline + ch*32);
          v16h am = aptr[(size_t)(ksbase + idx)*32];
          acc = __builtin_amdgcn_wmma_f32_16x16x32_f16(
              /*neg_a=*/false, am, /*neg_b=*/false, bm,
              /*c_mod=*/(short)0, acc, /*reuse_a=*/false, /*reuse_b=*/false);
        }
      }
    }
  }

  // D layout: VGPR r -> M=r (lanes 0-15), M=r+8 (lanes 16-31); N = lane&15
  int wcol  = nt*16 + n;
  int mbase = mt*16 + ((lane < 16) ? 0 : 8);
  float* ybase = y_conv + (((size_t)b*COUT + mbase)*DD + d)*HH*WW
               + (size_t)h*WW + wcol;
  #pragma unroll
  for (int r = 0; r < 8; ++r)
    ybase[(size_t)r*DHW] = acc[r];
}

// ---------------------------------------------------------------------------
// Kernel 5: fused trilinear 2x upsample + 1x1x1 voxelize.
//   One thread per output spatial voxel; corner weights shared across channels.
// ---------------------------------------------------------------------------
__global__ void upsample_vox_kernel(const float* __restrict__ y_conv,
                                    const float* __restrict__ vox_w,
                                    const float* __restrict__ vox_b,
                                    float* __restrict__ out) {
  int b  = blockIdx.y;
  int dz = blockIdx.x >> 6;
  int dy = blockIdx.x & 63;
  int dx = threadIdx.x;                   // 64 threads

  int z0, z1, y0, y1, x0, x1;
  float wz, wy, wx;
  {
    float s = fmaxf((dz + 0.5f)*0.5f - 0.5f, 0.0f);
    z0 = (int)floorf(s); z1 = min(z0 + 1, 31); wz = s - (float)z0;
    s = fmaxf((dy + 0.5f)*0.5f - 0.5f, 0.0f);
    y0 = (int)floorf(s); y1 = min(y0 + 1, 31); wy = s - (float)y0;
    s = fmaxf((dx + 0.5f)*0.5f - 0.5f, 0.0f);
    x0 = (int)floorf(s); x1 = min(x0 + 1, 31); wx = s - (float)x0;
  }
  int o000 = z0*1024 + y0*32 + x0, o001 = z0*1024 + y0*32 + x1;
  int o010 = z0*1024 + y1*32 + x0, o011 = z0*1024 + y1*32 + x1;
  int o100 = z1*1024 + y0*32 + x0, o101 = z1*1024 + y0*32 + x1;
  int o110 = z1*1024 + y1*32 + x0, o111 = z1*1024 + y1*32 + x1;

  const float* src = y_conv + (size_t)b*COUT*DHW;
  float* ydst = out;
  float* odst = out + (size_t)BB*COUT*UPVOX;
  size_t spix = (size_t)dz*UP*UP + dy*UP + dx;

  float vacc = vox_b[0];
  #pragma unroll 4
  for (int c = 0; c < COUT; ++c) {
    const float* p = src + (size_t)c*DHW;
    float c00 = p[o000]*(1.0f-wx) + p[o001]*wx;
    float c01 = p[o010]*(1.0f-wx) + p[o011]*wx;
    float c10 = p[o100]*(1.0f-wx) + p[o101]*wx;
    float c11 = p[o110]*(1.0f-wx) + p[o111]*wx;
    float c0  = c00*(1.0f-wy) + c01*wy;
    float c1  = c10*(1.0f-wy) + c11*wy;
    float val = c0*(1.0f-wz) + c1*wz;
    ydst[((size_t)b*COUT + c)*UPVOX + spix] = val;
    vacc += vox_w[c]*val;
  }
  odst[(size_t)b*UPVOX + spix] = vacc;
}

// ---------------------------------------------------------------------------
// Launch
// ---------------------------------------------------------------------------
extern "C" void kernel_launch(void* const* d_in, const int* in_sizes, int n_in,
                              void* d_out, int out_size, void* d_ws, size_t ws_size,
                              hipStream_t stream) {
  const float* x     = (const float*)d_in[0];
  const float* style = (const float*)d_in[1];
  const float* bank  = (const float*)d_in[2];
  const float* aff_w = (const float*)d_in[3];
  const float* aff_b = (const float*)d_in[4];
  const float* sel_w = (const float*)d_in[5];
  const float* sel_b = (const float*)d_in[6];
  const float* vox_w = (const float*)d_in[7];
  const float* vox_b = (const float*)d_in[8];

  // workspace layout
  float*     ws_s     = (float*)d_ws;                    // B*CIN    = 256 f32
  float*     ws_alpha = ws_s + BB*CIN;                   // B*BANK   = 16 f32
  float*     ws_demod = ws_alpha + BB*BANK;              // B*COUT   = 128 f32
  _Float16*  ws_afrag = (_Float16*)(ws_demod + BB*COUT); // B*2*54*32*16 halfs (32B aligned)
  float*     ws_yconv = (float*)(ws_afrag + (size_t)BB*2*NKS*32*16); // B*COUT*DHW f32

  prep_stats_kernel<<<BB, 128, 0, stream>>>(style, sel_w, sel_b, aff_w, aff_b,
                                            ws_s, ws_alpha);
  prep_demod_kernel<<<BB*COUT, 128, 0, stream>>>(bank, ws_s, ws_alpha, ws_demod);
  prep_afrag_kernel<<<BB*2*NKS, 32, 0, stream>>>(bank, ws_s, ws_alpha, ws_demod,
                                                 ws_afrag);
  conv_wmma_kernel<<<dim3(DD*HH, BB), 128, 0, stream>>>(x, ws_afrag, ws_yconv);
  upsample_vox_kernel<<<dim3(UP*UP, BB), UP, 0, stream>>>(ws_yconv, vox_w, vox_b,
                                                          (float*)d_out);
}
